// Equi_gcn_21603685499523
// MI455X (gfx1250) — compile-verified, matrix operand order checked by reading, not run on hardware
//
#include <hip/hip_runtime.h>
#include <hip/hip_bf16.h>
#include <float.h>
#include <math.h>

// ---------------------------------------------------------------------------
// Types for CDNA5 WMMA (wave32, 16x16x32 bf16 -> f32)
// ---------------------------------------------------------------------------
typedef __attribute__((ext_vector_type(16))) __bf16          v16bf;
typedef __attribute__((ext_vector_type(16))) unsigned short  v16u;
typedef __attribute__((ext_vector_type(8)))  unsigned short  v8u;
typedef __attribute__((ext_vector_type(8)))  float           v8f;

static __device__ __forceinline__ unsigned short f32_bf16(float f) {
  unsigned int u = __float_as_uint(f);
  u += 0x7FFFu + ((u >> 16) & 1u);         // round-to-nearest-even
  return (unsigned short)(u >> 16);
}

// ---------------------------------------------------------------------------
// Brute-force kNN: one query per thread, candidates tiled through LDS.
// Maintains a sorted (KK+1)-entry best list; entry 0 is self, dropped on write.
// ---------------------------------------------------------------------------
template <int KK>
__global__ void knn_kernel(const float* __restrict__ v, int n, int nq, int qstride,
                           int* __restrict__ idx_out) {
  __shared__ float sp[128 * 3];
  const int b = blockIdx.y;
  const int q = blockIdx.x * blockDim.x + threadIdx.x;
  const bool act = (q < nq);
  const int qi = act ? q * qstride : 0;
  const float* vb = v + (size_t)b * n * 3;
  const float qx = vb[qi * 3 + 0], qy = vb[qi * 3 + 1], qz = vb[qi * 3 + 2];

  float bd[KK + 1];
  int   bi[KK + 1];
#pragma unroll
  for (int t = 0; t <= KK; ++t) { bd[t] = FLT_MAX; bi[t] = 0; }

  for (int t0 = 0; t0 < n; t0 += 128) {
    const int m = t0 + threadIdx.x;   // n is a multiple of 128
    sp[threadIdx.x * 3 + 0] = vb[m * 3 + 0];
    sp[threadIdx.x * 3 + 1] = vb[m * 3 + 1];
    sp[threadIdx.x * 3 + 2] = vb[m * 3 + 2];
    __syncthreads();
    if (act) {
      for (int j = 0; j < 128; ++j) {
        const float dx = sp[j * 3 + 0] - qx;
        const float dy = sp[j * 3 + 1] - qy;
        const float dz = sp[j * 3 + 2] - qz;
        const float d = dx * dx + dy * dy + dz * dz;
        if (d < bd[KK]) {
          bd[KK] = d; bi[KK] = t0 + j;
#pragma unroll
          for (int t = KK; t > 0; --t) {
            if (bd[t] < bd[t - 1]) {
              const float td = bd[t]; bd[t] = bd[t - 1]; bd[t - 1] = td;
              const int   ti = bi[t]; bi[t] = bi[t - 1]; bi[t - 1] = ti;
            }
          }
        }
      }
    }
    __syncthreads();
  }
  if (act) {
#pragma unroll
    for (int t = 0; t < KK; ++t)
      idx_out[((size_t)b * nq + q) * KK + t] = bi[t + 1];   // drop self
  }
}

// ---------------------------------------------------------------------------
// Pack weights: Wt_bf16[o, c] = bf16(W[c, o])  (A-operand friendly layout)
// ---------------------------------------------------------------------------
__global__ void pack_w_kernel(const float* __restrict__ W, unsigned short* __restrict__ Wt,
                              int cin, int cout) {
  const int tid = blockIdx.x * blockDim.x + threadIdx.x;
  if (tid >= cin * cout) return;
  const int c = tid % cin;
  const int o = tid / cin;
  Wt[tid] = f32_bf16(W[(size_t)c * cout + o]);
}

// ---------------------------------------------------------------------------
// conv_surface: mx[a] = max_k relu(unitdir(n,k) . K0[c,a]).
// Writes packed bf16 fm0 (b, n, a, c) for the WMMA layer AND fea_of out0.
// ---------------------------------------------------------------------------
__global__ void conv_surface_kernel(const float* __restrict__ v, const int* __restrict__ idx,
                                    const float* __restrict__ K0,
                                    unsigned short* __restrict__ fm0_bf,
                                    float* __restrict__ out0, int n, int Kn) {
  const int tid = blockIdx.x * blockDim.x + threadIdx.x;
  const int total = 2 * n * 32;
  if (tid >= total) return;
  const int c  = tid & 31;
  const int n0 = (tid >> 5) % n;
  const int b  = tid / (n * 32);
  const float* vb = v + (size_t)b * n * 3;
  const float px = vb[n0 * 3 + 0], py = vb[n0 * 3 + 1], pz = vb[n0 * 3 + 2];
  float mx[12];
#pragma unroll
  for (int a = 0; a < 12; ++a) mx[a] = 0.f;
  for (int k = 0; k < Kn; ++k) {
    const int nb = idx[((size_t)b * n + n0) * Kn + k];
    float dx = vb[nb * 3 + 0] - px, dy = vb[nb * 3 + 1] - py, dz = vb[nb * 3 + 2] - pz;
    const float inv = 1.f / (sqrtf(dx * dx + dy * dy + dz * dz) + 1e-8f);
    dx *= inv; dy *= inv; dz *= inv;
#pragma unroll
    for (int a = 0; a < 12; ++a) {
      const float* kk = K0 + ((size_t)c * 12 + a) * 3;
      const float th = fmaxf(dx * kk[0] + dy * kk[1] + dz * kk[2], 0.f);
      mx[a] = fmaxf(mx[a], th);
    }
  }
  float fmax_a = -FLT_MAX;
#pragma unroll
  for (int a = 0; a < 12; ++a) {
    fm0_bf[(((size_t)b * n + n0) * 12 + a) * 32 + c] = f32_bf16(mx[a]);
    fmax_a = fmaxf(fmax_a, mx[a]);
  }
  out0[((size_t)b * n + n0) * 32 + c] = fmax_a;
}

// ---------------------------------------------------------------------------
// theta[b,n,k,a] = relu(unitdir(n,k) . dirs[a])
// ---------------------------------------------------------------------------
__global__ void theta_kernel(const float* __restrict__ v, const int* __restrict__ idx,
                             const float* __restrict__ dirs, float* __restrict__ theta,
                             int n, int Kn) {
  const int tid = blockIdx.x * blockDim.x + threadIdx.x;
  const int total = 2 * n * Kn;
  if (tid >= total) return;
  const int k  = tid % Kn;
  const int n0 = (tid / Kn) % n;
  const int b  = tid / (n * Kn);
  const float* vb = v + (size_t)b * n * 3;
  const int nb = idx[((size_t)b * n + n0) * Kn + k];
  float dx = vb[nb * 3 + 0] - vb[n0 * 3 + 0];
  float dy = vb[nb * 3 + 1] - vb[n0 * 3 + 1];
  float dz = vb[nb * 3 + 2] - vb[n0 * 3 + 2];
  const float inv = 1.f / (sqrtf(dx * dx + dy * dy + dz * dz) + 1e-8f);
  dx *= inv; dy *= inv; dz *= inv;
  float* o = theta + (size_t)tid * 12;
#pragma unroll
  for (int a = 0; a < 12; ++a)
    o[a] = fmaxf(dx * dirs[a * 3 + 0] + dy * dirs[a * 3 + 1] + dz * dirs[a * 3 + 2], 0.f);
}

// ---------------------------------------------------------------------------
// equi_conv via v_wmma_f32_16x16x32_bf16, operands pre-packed in memory.
// fm_bf: (b, n, a, c) bf16;  Wst/Wct: (o, c) bf16.
// A lane's 32-deep K-chunk fragment = channels [hiK, hiK+8) U [hiK+16, hiK+24)
// -> exactly two 16-byte loads per fragment, zero conversion ALU.
// 4 waves/block; each wave owns MBLK 16-cout tiles (M register blocking), so
// every B fragment load feeds MBLK WMMAs (loads:wmma = 2:MBLK per chunk).
// ---------------------------------------------------------------------------
template <int CIN32, int MBLK>
__launch_bounds__(128)
__global__ void equi_conv_kernel(const unsigned short* __restrict__ fm_bf,
                                 const int* __restrict__ idx,
                                 const float* __restrict__ theta,
                                 const unsigned short* __restrict__ Wst,
                                 const unsigned short* __restrict__ Wct,
                                 const float* __restrict__ bc,
                                 float* __restrict__ fm_out,
                                 int n, int Kn, int cout) {
  const int lane  = threadIdx.x & 31;
  const int wave  = threadIdx.x >> 5;
  const int colM  = lane & 15;             // A: row M;  B: col N;  C/D: col N
  const int hiK   = (lane & 16) >> 1;      // 0 or 8
  const int b     = blockIdx.z;
  const int obase = (blockIdx.x * 4 + wave) * (MBLK * 16);
  const int p     = blockIdx.y * 16 + colM;   // flattened (n0, a0)
  const int n0    = p / 12;
  const int a0    = p - n0 * 12;
  const int cin   = CIN32 * 32;

  auto frag = [&](const unsigned short* base) -> v16bf {
    const v8u lo = *(const v8u*)(base + hiK);
    const v8u hi = *(const v8u*)(base + hiK + 16);
    v16u u;
#pragma unroll
    for (int h = 0; h < 8; ++h) { u[h] = lo[h]; u[h + 8] = hi[h]; }
    return __builtin_bit_cast(v16bf, u);
  };

  // Persistent A tiles for Ws, MBLK cout-tiles per wave
  v16bf aWs[MBLK][CIN32];
#pragma unroll
  for (int m = 0; m < MBLK; ++m) {
    const unsigned short* wsrow = Wst + (size_t)(obase + m * 16 + colM) * cin;
#pragma unroll
    for (int cc = 0; cc < CIN32; ++cc) aWs[m][cc] = frag(wsrow + cc * 32);
  }

  // fc = Wc^T . fm(center): B fragment shared across the MBLK tiles
  const unsigned short* fmc = fm_bf + ((size_t)(b * n + n0) * 12 + a0) * cin;
  v8f fc[MBLK];
#pragma unroll
  for (int m = 0; m < MBLK; ++m) fc[m] = (v8f){};
#pragma unroll
  for (int cc = 0; cc < CIN32; ++cc) {
    const v16bf bfrag = frag(fmc + cc * 32);
#pragma unroll
    for (int m = 0; m < MBLK; ++m) {
      const unsigned short* wcrow = Wct + (size_t)(obase + m * 16 + colM) * cin;
      fc[m] = __builtin_amdgcn_wmma_f32_16x16x32_bf16(
          false, frag(wcrow + cc * 32), false, bfrag, (short)0, fc[m], false, false);
    }
  }

  // fs = max_k theta_k * (Ws^T . fm(neighbor_k))
  v8f best[MBLK];
#pragma unroll
  for (int m = 0; m < MBLK; ++m)
#pragma unroll
    for (int r = 0; r < 8; ++r) best[m][r] = -FLT_MAX;

  for (int k = 0; k < Kn; ++k) {
    const int   nb = idx[((size_t)b * n + n0) * Kn + k];
    const float th = theta[(((size_t)b * n + n0) * Kn + k) * 12 + a0];
    const unsigned short* fmn = fm_bf + ((size_t)(b * n + nb) * 12 + a0) * cin;
    v8f acc[MBLK];
#pragma unroll
    for (int m = 0; m < MBLK; ++m) acc[m] = (v8f){};
#pragma unroll
    for (int cc = 0; cc < CIN32; ++cc) {
      const v16bf bfrag = frag(fmn + cc * 32);
#pragma unroll
      for (int m = 0; m < MBLK; ++m) {
        acc[m] = __builtin_amdgcn_wmma_f32_16x16x32_bf16(
            false, aWs[m][cc], false, bfrag, (short)0, acc[m], false, false);
      }
    }
#pragma unroll
    for (int m = 0; m < MBLK; ++m)
#pragma unroll
      for (int r = 0; r < 8; ++r) best[m][r] = fmaxf(best[m][r], acc[m][r] * th);
  }

  // C/D layout: lane -> column p; VGPR r -> row (r + 8*(lane>=16))
  const int mbase = (lane >> 4) * 8;
  const size_t out_b = (size_t)b * cout * n * 12;
#pragma unroll
  for (int m = 0; m < MBLK; ++m)
#pragma unroll
    for (int r = 0; r < 8; ++r) {
      const int o = obase + m * 16 + mbase + r;
      fm_out[out_b + ((size_t)o * n + n0) * 12 + a0] = fc[m][r] + bc[o] + best[m][r];
    }
}

// ---------------------------------------------------------------------------
// BatchNorm over (b, n, a) per channel: stats then apply+relu (in place),
// optionally emitting the packed bf16 (b, n, a, c) copy for the next layer.
// ---------------------------------------------------------------------------
__global__ void bn_stats_kernel(const float* __restrict__ x, float* __restrict__ mv,
                                int C, int nA) {
  __shared__ float ss[256], sq[256];
  const int c = blockIdx.x;
  float s = 0.f, s2 = 0.f;
  const int tot = 2 * nA;
  for (int t = threadIdx.x; t < tot; t += 256) {
    const int b = t / nA, r = t - b * nA;
    const float xv = x[((size_t)b * C + c) * nA + r];
    s += xv; s2 += xv * xv;
  }
  ss[threadIdx.x] = s; sq[threadIdx.x] = s2;
  __syncthreads();
  for (int off = 128; off > 0; off >>= 1) {
    if (threadIdx.x < off) {
      ss[threadIdx.x] += ss[threadIdx.x + off];
      sq[threadIdx.x] += sq[threadIdx.x + off];
    }
    __syncthreads();
  }
  if (threadIdx.x == 0) {
    const float m = ss[0] / (float)tot;
    mv[c]     = m;
    mv[C + c] = sq[0] / (float)tot - m * m;
  }
}

__global__ void bn_relu_kernel(float* __restrict__ x, const float* __restrict__ mv,
                               const float* __restrict__ g, const float* __restrict__ bb,
                               unsigned short* __restrict__ pack, int C, int nA) {
  const int tid = blockIdx.x * blockDim.x + threadIdx.x;
  const int total = 2 * C * nA;
  if (tid >= total) return;
  const int c = (tid / nA) % C;
  const float y = fmaxf((x[tid] - mv[c]) * rsqrtf(mv[C + c] + 1e-5f) * g[c] + bb[c], 0.f);
  x[tid] = y;
  if (pack) {
    const int b = tid / (C * nA);
    const int r = tid % nA;
    pack[((size_t)b * nA + r) * C + c] = f32_bf16(y);
  }
}

// ---------------------------------------------------------------------------
// Pool: bf16-packed out[(b,i,a),c] = max(fm[.,4i,.], max_{j<4} fm[., idxp, .])
// ---------------------------------------------------------------------------
__global__ void pool_kernel(const float* __restrict__ fm, const int* __restrict__ idxp,
                            unsigned short* __restrict__ out_bf, int n, int nq, int C) {
  const int tid = blockIdx.x * blockDim.x + threadIdx.x;
  const int total = 2 * C * nq;
  if (tid >= total) return;
  const int i = tid % nq;
  const int c = (tid / nq) % C;
  const int b = tid / (nq * C);
  const float* base = fm + (((size_t)b * C + c) * n) * 12;
  int nb[4];
#pragma unroll
  for (int j = 0; j < 4; ++j) nb[j] = idxp[((size_t)b * nq + i) * 4 + j];
#pragma unroll
  for (int a = 0; a < 12; ++a) {
    float mx = base[(size_t)(4 * i) * 12 + a];
#pragma unroll
    for (int j = 0; j < 4; ++j) mx = fmaxf(mx, base[(size_t)nb[j] * 12 + a]);
    out_bf[(((size_t)b * nq + i) * 12 + a) * C + c] = f32_bf16(mx);
  }
}

__global__ void subsample_v_kernel(const float* __restrict__ v, float* __restrict__ vo,
                                   int n, int nq) {
  const int tid = blockIdx.x * blockDim.x + threadIdx.x;
  const int total = 2 * nq * 3;
  if (tid >= total) return;
  const int b = tid / (nq * 3);
  const int r = tid - b * nq * 3;
  const int i = r / 3, d = r - i * 3;
  vo[tid] = v[(size_t)b * n * 3 + (size_t)(i * 4) * 3 + d];
}

// ---------------------------------------------------------------------------
// fea_of: out[b,n,c] = max_a fm[b,c,n,a]
// ---------------------------------------------------------------------------
__global__ void fea_of_kernel(const float* __restrict__ fm, float* __restrict__ out,
                              int n, int C) {
  const int tid = blockIdx.x * blockDim.x + threadIdx.x;
  const int total = 2 * n * C;
  if (tid >= total) return;
  const int c  = tid % C;
  const int n0 = (tid / C) % n;
  const int b  = tid / (n * C);
  const float* p = fm + (((size_t)b * C + c) * n + n0) * 12;
  float mx = -FLT_MAX;
#pragma unroll
  for (int a = 0; a < 12; ++a) mx = fmaxf(mx, p[a]);
  out[tid] = mx;   // tid == (b*n + n0)*C + c  -> (b, n, c)
}

// ---------------------------------------------------------------------------
// Orchestration
// ---------------------------------------------------------------------------
extern "C" void kernel_launch(void* const* d_in, const int* in_sizes, int n_in,
                              void* d_out, int out_size, void* d_ws, size_t ws_size,
                              hipStream_t stream) {
  const float* v     = (const float*)d_in[0];
  const float* K0    = (const float*)d_in[1];
  const float* dirs1 = (const float*)d_in[2];
  const float* Wc1   = (const float*)d_in[3];
  const float* bc1   = (const float*)d_in[4];
  const float* Ws1   = (const float*)d_in[5];
  const float* dirs2 = (const float*)d_in[6];
  const float* Wc2   = (const float*)d_in[7];
  const float* bc2   = (const float*)d_in[8];
  const float* Ws2   = (const float*)d_in[9];
  const float* dirs3 = (const float*)d_in[10];
  const float* Wc3   = (const float*)d_in[11];
  const float* bc3   = (const float*)d_in[12];
  const float* Ws3   = (const float*)d_in[13];
  const float* dirs4 = (const float*)d_in[14];
  const float* Wc4   = (const float*)d_in[15];
  const float* bc4   = (const float*)d_in[16];
  const float* Ws4   = (const float*)d_in[17];
  const float* g1    = (const float*)d_in[18];
  const float* b1    = (const float*)d_in[19];
  const float* g2    = (const float*)d_in[20];
  const float* b2    = (const float*)d_in[21];
  const float* g3    = (const float*)d_in[22];
  const float* b3    = (const float*)d_in[23];
  float* out = (float*)d_out;

  char* w = (char*)d_ws;
  auto alloc = [&](size_t bytes) -> char* {
    char* p = w;
    w += (bytes + 255) & ~(size_t)255;
    return p;
  };
  const int B = 2, N = 4096, Kn = 20, A_ = 12;
  int*            idx0    = (int*)           alloc((size_t)B * N * Kn * 4);
  unsigned short* fm0bf   = (unsigned short*)alloc((size_t)B * N * A_ * 32 * 2);
  float*          theta1  = (float*)         alloc((size_t)B * N * Kn * A_ * 4);
  float*          fm1     = (float*)         alloc((size_t)B * 64 * N * A_ * 4);
  int*            idxp1   = (int*)           alloc((size_t)B * 1024 * 4 * 4);
  float*          v1      = (float*)         alloc((size_t)B * 1024 * 3 * 4);
  unsigned short* fmp1bf  = (unsigned short*)alloc((size_t)B * 1024 * A_ * 64 * 2);
  int*            idx1    = (int*)           alloc((size_t)B * 1024 * Kn * 4);
  float*          theta2  = (float*)         alloc((size_t)B * 1024 * Kn * A_ * 4);
  float*          fm2     = (float*)         alloc((size_t)B * 128 * 1024 * A_ * 4);
  unsigned short* fm2bf   = (unsigned short*)alloc((size_t)B * 1024 * A_ * 128 * 2);
  float*          theta3  = (float*)         alloc((size_t)B * 1024 * Kn * A_ * 4);
  float*          fm3     = (float*)         alloc((size_t)B * 256 * 1024 * A_ * 4);
  int*            idxp2   = (int*)           alloc((size_t)B * 256 * 4 * 4);
  float*          v2      = (float*)         alloc((size_t)B * 256 * 3 * 4);
  unsigned short* fmp2bf  = (unsigned short*)alloc((size_t)B * 256 * A_ * 256 * 2);
  int*            idx2    = (int*)           alloc((size_t)B * 256 * Kn * 4);
  float*          theta4  = (float*)         alloc((size_t)B * 256 * Kn * A_ * 4);
  float*          fm4     = (float*)         alloc((size_t)B * 512 * 256 * A_ * 4);
  float*          mv      = (float*)         alloc(1024 * 4);
  unsigned short* Ws1t    = (unsigned short*)alloc((size_t)32  * 64  * 2);
  unsigned short* Wc1t    = (unsigned short*)alloc((size_t)32  * 64  * 2);
  unsigned short* Ws2t    = (unsigned short*)alloc((size_t)64  * 128 * 2);
  unsigned short* Wc2t    = (unsigned short*)alloc((size_t)64  * 128 * 2);
  unsigned short* Ws3t    = (unsigned short*)alloc((size_t)128 * 256 * 2);
  unsigned short* Wc3t    = (unsigned short*)alloc((size_t)128 * 256 * 2);
  unsigned short* Ws4t    = (unsigned short*)alloc((size_t)256 * 512 * 2);
  unsigned short* Wc4t    = (unsigned short*)alloc((size_t)256 * 512 * 2);

  float* out0 = out;                 // (2,4096,32)
  float* out1 = out + 262144;        // (2,4096,64)
  float* out2 = out + 786432;        // (2,1024,128)
  float* out3 = out + 1048576;       // (2,1024,256)
  float* out4 = out + 1572864;       // (2,256,512)

  // --- weight packing (tiny, L2-resident) ---
  pack_w_kernel<<<(32 * 64 + 255) / 256, 256, 0, stream>>>(Ws1, Ws1t, 32, 64);
  pack_w_kernel<<<(32 * 64 + 255) / 256, 256, 0, stream>>>(Wc1, Wc1t, 32, 64);
  pack_w_kernel<<<(64 * 128 + 255) / 256, 256, 0, stream>>>(Ws2, Ws2t, 64, 128);
  pack_w_kernel<<<(64 * 128 + 255) / 256, 256, 0, stream>>>(Wc2, Wc2t, 64, 128);
  pack_w_kernel<<<(128 * 256 + 255) / 256, 256, 0, stream>>>(Ws3, Ws3t, 128, 256);
  pack_w_kernel<<<(128 * 256 + 255) / 256, 256, 0, stream>>>(Wc3, Wc3t, 128, 256);
  pack_w_kernel<<<(256 * 512 + 255) / 256, 256, 0, stream>>>(Ws4, Ws4t, 256, 512);
  pack_w_kernel<<<(256 * 512 + 255) / 256, 256, 0, stream>>>(Wc4, Wc4t, 256, 512);

  // --- stage 0: graph + surface conv (emits packed fm0 + out0) ---
  knn_kernel<20><<<dim3(N / 128, B), 128, 0, stream>>>(v, N, N, 1, idx0);
  conv_surface_kernel<<<(B * N * 32 + 255) / 256, 256, 0, stream>>>(v, idx0, K0, fm0bf, out0, N, Kn);

  // --- layer 1: 32 -> 64 @ n=4096 ---
  theta_kernel<<<(B * N * Kn + 255) / 256, 256, 0, stream>>>(v, idx0, dirs1, theta1, N, Kn);
  equi_conv_kernel<1, 1><<<dim3(1, N * 12 / 16, B), 128, 0, stream>>>(
      fm0bf, idx0, theta1, Ws1t, Wc1t, bc1, fm1, N, Kn, 64);
  bn_stats_kernel<<<64, 256, 0, stream>>>(fm1, mv, 64, N * A_);
  bn_relu_kernel<<<(B * 64 * N * A_ + 255) / 256, 256, 0, stream>>>(
      fm1, mv, g1, b1, (unsigned short*)nullptr, 64, N * A_);
  fea_of_kernel<<<(B * N * 64 + 255) / 256, 256, 0, stream>>>(fm1, out1, N, 64);

  // --- pool 4096 -> 1024 (emits packed fmp1) ---
  knn_kernel<4><<<dim3(1024 / 128, B), 128, 0, stream>>>(v, N, 1024, 4, idxp1);
  pool_kernel<<<(B * 64 * 1024 + 255) / 256, 256, 0, stream>>>(fm1, idxp1, fmp1bf, N, 1024, 64);
  subsample_v_kernel<<<(B * 1024 * 3 + 255) / 256, 256, 0, stream>>>(v, v1, N, 1024);
  knn_kernel<20><<<dim3(1024 / 128, B), 128, 0, stream>>>(v1, 1024, 1024, 1, idx1);

  // --- layer 2: 64 -> 128 @ n=1024 (bn_relu also packs fm2 for layer 3) ---
  theta_kernel<<<(B * 1024 * Kn + 255) / 256, 256, 0, stream>>>(v1, idx1, dirs2, theta2, 1024, Kn);
  equi_conv_kernel<2, 2><<<dim3(1, 1024 * 12 / 16, B), 128, 0, stream>>>(
      fmp1bf, idx1, theta2, Ws2t, Wc2t, bc2, fm2, 1024, Kn, 128);
  bn_stats_kernel<<<128, 256, 0, stream>>>(fm2, mv, 128, 1024 * A_);
  bn_relu_kernel<<<(B * 128 * 1024 * A_ + 255) / 256, 256, 0, stream>>>(
      fm2, mv, g2, b2, fm2bf, 128, 1024 * A_);
  fea_of_kernel<<<(B * 1024 * 128 + 255) / 256, 256, 0, stream>>>(fm2, out2, 1024, 128);

  // --- layer 3: 128 -> 256 @ n=1024 ---
  theta_kernel<<<(B * 1024 * Kn + 255) / 256, 256, 0, stream>>>(v1, idx1, dirs3, theta3, 1024, Kn);
  equi_conv_kernel<4, 2><<<dim3(2, 1024 * 12 / 16, B), 128, 0, stream>>>(
      fm2bf, idx1, theta3, Ws3t, Wc3t, bc3, fm3, 1024, Kn, 256);
  bn_stats_kernel<<<256, 256, 0, stream>>>(fm3, mv, 256, 1024 * A_);
  bn_relu_kernel<<<(B * 256 * 1024 * A_ + 255) / 256, 256, 0, stream>>>(
      fm3, mv, g3, b3, (unsigned short*)nullptr, 256, 1024 * A_);
  fea_of_kernel<<<(B * 1024 * 256 + 255) / 256, 256, 0, stream>>>(fm3, out3, 1024, 256);

  // --- pool 1024 -> 256 (emits packed fmp2) ---
  knn_kernel<4><<<dim3(256 / 128, B), 128, 0, stream>>>(v1, 1024, 256, 4, idxp2);
  pool_kernel<<<(B * 256 * 256 + 255) / 256, 256, 0, stream>>>(fm3, idxp2, fmp2bf, 1024, 256, 256);
  subsample_v_kernel<<<(B * 256 * 3 + 255) / 256, 256, 0, stream>>>(v1, v2, 1024, 256);
  knn_kernel<20><<<dim3(256 / 128, B), 128, 0, stream>>>(v2, 256, 256, 1, idx2);

  // --- layer 4: 256 -> 512 @ n=256 (no BN/relu) ---
  theta_kernel<<<(B * 256 * Kn + 255) / 256, 256, 0, stream>>>(v2, idx2, dirs4, theta4, 256, Kn);
  equi_conv_kernel<8, 2><<<dim3(4, 256 * 12 / 16, B), 128, 0, stream>>>(
      fmp2bf, idx2, theta4, Ws4t, Wc4t, bc4, fm4, 256, Kn, 512);
  fea_of_kernel<<<(B * 256 * 512 + 255) / 256, 256, 0, stream>>>(fm4, out4, 256, 512);

  (void)in_sizes; (void)n_in; (void)out_size; (void)ws_size;
}
